// MPNN_34076270526640
// MI455X (gfx1250) — matrix operand order repaired and checked by reading, not compiled, not run on hardware
//
#include <hip/hip_runtime.h>

// ---------------------------------------------------------------------------
// MPNN on MI455X (gfx1250, wave32).
//
//  m_e = e1_e @ P_src + Q_src,  P = h @ B_ext  (WMMA f16 GEMM, K=32 exact)
//  P f16 (105 MB) -> edge gathers served from 192 MB L2, not HBM.
//  Node update (NNConv root + GRU) also runs on WMMA via pre-swizzled
//  f16 fragment tables + per-wave LDS transpose for the u operand.
// ---------------------------------------------------------------------------

#define NN     50000
#define EE     250000
#define FIN    64
#define DD     32
#define NSTEPS 3
#define PCOLS  1056             // 1024 (h,o) cols + 32 be2 cols folded in
#define NTILES (NN / 16)        // 3125 exact
#define CTILES (PCOLS / 16)     // 66

typedef _Float16 h16;
typedef _Float16 v16h __attribute__((ext_vector_type(16)));
typedef _Float16 v8h  __attribute__((ext_vector_type(8)));
typedef _Float16 v4h  __attribute__((ext_vector_type(4)));
typedef float    v8f  __attribute__((ext_vector_type(8)));

__device__ __forceinline__ float lane_bcast(float v, int l) {
  return __uint_as_float(__builtin_amdgcn_readlane(__float_as_uint(v), l));
}
__device__ __forceinline__ float sigmoidf_(float x) { return 1.0f / (1.0f + expf(-x)); }
__device__ __forceinline__ v8f wmma16(v16h a, v16h b, v8f c) {
  return __builtin_amdgcn_wmma_f32_16x16x32_f16(false, a, false, b, (short)0, c, false, false);
}

// ---------------------------------------------------------------------------
// Setup: all right-hand matrices stored as pre-swizzled WMMA B-fragments:
//   frag[tile][lane][e] = B[K][col],  K = ((lane&16)>>1)+(e&7)+((e&8)<<1), col = tile*16 + (lane&15)
// so a B operand is one 32-byte contiguous per-lane load.
// ---------------------------------------------------------------------------
__global__ void k_prep(const float* __restrict__ We2, const float* __restrict__ be2,
                       const float* __restrict__ root,
                       const float* __restrict__ W_ih, const float* __restrict__ W_hh,
                       h16* __restrict__ BF, h16* __restrict__ rootF,
                       h16* __restrict__ WihF, h16* __restrict__ WhhF) {
  int tid = blockIdx.x * blockDim.x + threadIdx.x;
  int stride = gridDim.x * blockDim.x;
  // B_ext[K=i][col=h*32+o] = We2[h*1024 + i*32 + o];  col 1024+o -> be2[i*32+o]
  for (int idx = tid; idx < CTILES * 512; idx += stride) {
    int c = idx >> 9, r = idx & 511, l = r >> 4, e = r & 15;
    int K = ((l & 16) >> 1) + (e & 7) + ((e & 8) << 1);
    int col = c * 16 + (l & 15);
    float v = (col < 1024) ? We2[(col >> 5) * 1024 + K * 32 + (col & 31)]
                           : be2[K * 32 + (col - 1024)];
    BF[idx] = (h16)v;
  }
  for (int idx = tid; idx < 2 * 512; idx += stride) {   // root: B[K][j] = root[K*32+j]
    int c = idx >> 9, r = idx & 511, l = r >> 4, e = r & 15;
    int K = ((l & 16) >> 1) + (e & 7) + ((e & 8) << 1), col = c * 16 + (l & 15);
    rootF[idx] = (h16)root[K * 32 + col];
  }
  for (int idx = tid; idx < 6 * 512; idx += stride) {   // gates: B[K][j] = W[j*32+K]
    int c = idx >> 9, r = idx & 511, l = r >> 4, e = r & 15;
    int K = ((l & 16) >> 1) + (e & 7) + ((e & 8) << 1), col = c * 16 + (l & 15);
    WihF[idx] = (h16)W_ih[col * 32 + K];
    WhhF[idx] = (h16)W_hh[col * 32 + K];
  }
}

// --- in-degree counts (for mean aggregation) ---
__global__ void k_count(const int* __restrict__ ei, float* __restrict__ cnt) {
  int e = blockIdx.x * blockDim.x + threadIdx.x;
  if (e < EE) unsafeAtomicAdd(&cnt[ei[EE + e]], 1.0f);
}

// --- projection: h = relu(x @ Wp + bp), one wave per node (runs once) ---
__global__ void k_proj(const float* __restrict__ x, const float* __restrict__ Wp,
                       const float* __restrict__ bp, float* __restrict__ h) {
  int wave = (blockIdx.x * blockDim.x + threadIdx.x) >> 5;
  int lane = threadIdx.x & 31;
  if (wave >= NN) return;
  float x0 = x[wave * FIN + lane];
  float x1 = x[wave * FIN + 32 + lane];
  float acc = bp[lane];
#pragma unroll
  for (int i = 0; i < 32; ++i) acc += lane_bcast(x0, i) * Wp[i * 32 + lane];
#pragma unroll
  for (int i = 0; i < 32; ++i) acc += lane_bcast(x1, i) * Wp[(32 + i) * 32 + lane];
  h[wave * DD + lane] = fmaxf(acc, 0.0f);
}

// --- P = h @ B_ext via WMMA: one wave per 16-node tile, 66 col-tiles ---
__global__ void k_gemm(const float* __restrict__ h, const h16* __restrict__ BF,
                       h16* __restrict__ P) {
  int wave = (blockIdx.x * blockDim.x + threadIdx.x) >> 5;
  int lane = threadIdx.x & 31;
  if (wave >= NTILES) return;
  const int m  = lane & 15;
  const bool hi = lane >= 16;
  const int nb = wave * 16;

  // A fragment 16x32 f16: lanes 0-15 row m K{0..7,16..23}; lanes 16-31 K{8..15,24..31}
  const float* hrow = h + (size_t)(nb + m) * DD;
  const int k0 = hi ? 8 : 0;
  v16h a;
#pragma unroll
  for (int t = 0; t < 8; ++t) {
    a[t]     = (h16)hrow[k0 + t];
    a[8 + t] = (h16)hrow[k0 + 16 + t];
  }

  for (int c = 0; c < CTILES; ++c) {
    v16h b = *(const v16h*)(BF + (size_t)c * 512 + lane * 16);  // 2x b128, coalesced
    v8f acc = {};
    acc = wmma16(a, b, acc);
#pragma unroll
    for (int r = 0; r < 8; ++r) {
      int row = nb + (hi ? 8 + r : r);
      P[(size_t)row * PCOLS + c * 16 + m] = (h16)acc[r];
    }
  }
}

// --- edge pass: m_e = e1_e @ P_src + Q_src, atomic scatter to agg[dst] ---
// lanes: g = lane>>3 (h-group of 8), q = lane&7 (col quad). 8 iterations of
// one v4h (8B) load + bpermute broadcast + 4 mixed-precision FMAs, then a
// shfl_xor cross-group reduction and one f32 atomic per lane.
__global__ void k_edge(const int* __restrict__ ei, const float* __restrict__ eattr,
                       const float* __restrict__ We1, const float* __restrict__ be1,
                       const h16* __restrict__ P, float* __restrict__ agg) {
  int wave = (blockIdx.x * blockDim.x + threadIdx.x) >> 5;
  int lane = threadIdx.x & 31;
  if (wave >= EE) return;
  const int src = ei[wave];
  const int dst = ei[EE + wave];
  const float attr = eattr[wave];
  float e1 = fmaxf(attr * We1[lane] + be1[lane], 0.0f);   // lane holds e1[h=lane]

  const int g = lane >> 3;
  const int q = lane & 7;
  const h16* Prow = P + (size_t)src * PCOLS;
  float m0 = 0.f, m1 = 0.f, m2 = 0.f, m3 = 0.f;
#pragma unroll
  for (int t = 0; t < 8; ++t) {
    int hh = 4 * t + g;
    float eh = __shfl(e1, hh, 32);
    v4h p = *(const v4h*)(Prow + hh * 32 + 4 * q);
    m0 += eh * (float)p.x;
    m1 += eh * (float)p.y;
    m2 += eh * (float)p.z;
    m3 += eh * (float)p.w;
  }
  // reduce over the 4 h-groups (lanes l, l^8, l^16, l^24 share q)
  m0 += __shfl_xor(m0, 8, 32);  m0 += __shfl_xor(m0, 16, 32);
  m1 += __shfl_xor(m1, 8, 32);  m1 += __shfl_xor(m1, 16, 32);
  m2 += __shfl_xor(m2, 8, 32);  m2 += __shfl_xor(m2, 16, 32);
  m3 += __shfl_xor(m3, 8, 32);  m3 += __shfl_xor(m3, 16, 32);

  int col = 4 * q + g;                       // every lane owns one output col
  float mv = (g == 0) ? m0 : (g == 1) ? m1 : (g == 2) ? m2 : m3;
  mv += (float)Prow[1024 + col];             // be2 contribution
  unsafeAtomicAdd(&agg[(size_t)dst * DD + col], mv);
}

// --- node update on WMMA: conv (h@root) + ReLU + GRU gates, 16 nodes/wave ---
__global__ void __launch_bounds__(256)
k_node(float* __restrict__ h, const float* __restrict__ agg,
       const float* __restrict__ cnt, const h16* __restrict__ rootF,
       const float* __restrict__ conv_bias, const h16* __restrict__ WihF,
       const h16* __restrict__ WhhF, const float* __restrict__ b_ih,
       const float* __restrict__ b_hh) {
  __shared__ h16 uT[8][512];                 // per-wave 16x32 f16 tile for transpose
  int wave = (blockIdx.x * blockDim.x + threadIdx.x) >> 5;
  int wib  = (threadIdx.x >> 5) & 7;
  int lane = threadIdx.x & 31;
  if (wave >= NTILES) return;
  const int n  = lane & 15;
  const bool hi = lane >= 16;
  const int nb = wave * 16;
  const int k0 = hi ? 8 : 0;

  // h as A-fragment (f16)
  const float* hrow = h + (size_t)(nb + n) * DD;
  v16h hA;
#pragma unroll
  for (int t = 0; t < 8; ++t) {
    hA[t]     = (h16)hrow[k0 + t];
    hA[8 + t] = (h16)hrow[k0 + 16 + t];
  }

  // per-output-row reciprocal degree (D-layout rows)
  float rd[8];
#pragma unroll
  for (int r = 0; r < 8; ++r) {
    int row = nb + (hi ? 8 + r : r);
    rd[r] = __builtin_amdgcn_rcpf(fmaxf(cnt[row], 1.0f));
  }

  // conv: u = relu(agg/denom + h@root + conv_bias); C operand = agg/denom + bias
  v8f u[2];
  float hD[16];
#pragma unroll
  for (int c = 0; c < 2; ++c) {
    v8f cc;
#pragma unroll
    for (int r = 0; r < 8; ++r) {
      int row = nb + (hi ? 8 + r : r);
      int col = c * 16 + n;
      hD[c * 8 + r] = h[(size_t)row * DD + col];
      cc[r] = agg[(size_t)row * DD + col] * rd[r] + conv_bias[col];
    }
    v16h b = *(const v16h*)(rootF + (size_t)c * 512 + lane * 16);
    u[c] = wmma16(hA, b, cc);
#pragma unroll
    for (int r = 0; r < 8; ++r) u[c][r] = fmaxf(u[c][r], 0.0f);
  }

  // transpose u (D-layout) -> A-fragment via this wave's LDS slice.
  // Same-wave LDS ops are in-order; no barrier needed (disjoint slices).
  h16* myT = &uT[wib][0];
#pragma unroll
  for (int c = 0; c < 2; ++c)
#pragma unroll
    for (int r = 0; r < 8; ++r) {
      int row = hi ? 8 + r : r;
      myT[row * 32 + c * 16 + n] = (h16)u[c][r];
    }
  const h16* rr = myT + n * 32;              // A row = lane&15
  v8h lo = *(const v8h*)(rr + k0);
  v8h hi8 = *(const v8h*)(rr + k0 + 16);
  v16h uA;
#pragma unroll
  for (int t = 0; t < 8; ++t) { uA[t] = lo[t]; uA[8 + t] = hi8[t]; }

  // GRU gates: gi = u@W_ih^T + b_ih, gh = h@W_hh^T + b_hh  (6 col-tiles each)
  v8f gi[6], gh[6];
#pragma unroll
  for (int c = 0; c < 6; ++c) {
    float bv = b_ih[c * 16 + n];
    v8f cc;
#pragma unroll
    for (int r = 0; r < 8; ++r) cc[r] = bv;
    v16h b = *(const v16h*)(WihF + (size_t)c * 512 + lane * 16);
    gi[c] = wmma16(uA, b, cc);
  }
#pragma unroll
  for (int c = 0; c < 6; ++c) {
    float bv = b_hh[c * 16 + n];
    v8f cc;
#pragma unroll
    for (int r = 0; r < 8; ++r) cc[r] = bv;
    v16h b = *(const v16h*)(WhhF + (size_t)c * 512 + lane * 16);
    gh[c] = wmma16(hA, b, cc);
  }

  // combine gates elementwise in D-layout: tiles {c,2+c,4+c} share (row,col)
#pragma unroll
  for (int c = 0; c < 2; ++c)
#pragma unroll
    for (int r = 0; r < 8; ++r) {
      float rg = sigmoidf_(gi[c][r] + gh[c][r]);
      float zg = sigmoidf_(gi[2 + c][r] + gh[2 + c][r]);
      float ng = tanhf(gi[4 + c][r] + rg * gh[4 + c][r]);
      float hn = (1.0f - zg) * ng + zg * hD[c * 8 + r];
      int row = nb + (hi ? 8 + r : r);
      h[(size_t)row * DD + c * 16 + n] = hn;   // out == msg == h'
    }
}

extern "C" void kernel_launch(void* const* d_in, const int* in_sizes, int n_in,
                              void* d_out, int out_size, void* d_ws, size_t ws_size,
                              hipStream_t stream) {
  const float* x     = (const float*)d_in[0];
  const int*   ei    = (const int*)  d_in[1];
  const float* eattr = (const float*)d_in[2];
  const float* Wp    = (const float*)d_in[3];
  const float* bp    = (const float*)d_in[4];
  const float* We1   = (const float*)d_in[5];
  const float* be1   = (const float*)d_in[6];
  const float* We2   = (const float*)d_in[7];
  const float* be2   = (const float*)d_in[8];
  const float* root  = (const float*)d_in[9];
  const float* cbias = (const float*)d_in[10];
  const float* W_ih  = (const float*)d_in[11];
  const float* W_hh  = (const float*)d_in[12];
  const float* b_ih  = (const float*)d_in[13];
  const float* b_hh  = (const float*)d_in[14];

  char* ws = (char*)d_ws;
  auto al = [](size_t v) { return (v + 255) & ~(size_t)255; };
  size_t o = 0;
  h16*   BF    = (h16*)(ws + o);   o = al(o + (size_t)CTILES * 512 * sizeof(h16));
  h16*   rootF = (h16*)(ws + o);   o = al(o + (size_t)2 * 512 * sizeof(h16));
  h16*   WihF  = (h16*)(ws + o);   o = al(o + (size_t)6 * 512 * sizeof(h16));
  h16*   WhhF  = (h16*)(ws + o);   o = al(o + (size_t)6 * 512 * sizeof(h16));
  float* cnt   = (float*)(ws + o); o = al(o + (size_t)NN * sizeof(float));
  float* agg   = (float*)(ws + o); o = al(o + (size_t)NN * DD * sizeof(float));
  h16*   P     = (h16*)(ws + o);   o = al(o + (size_t)NN * PCOLS * sizeof(h16));
  (void)ws_size;

  float* h = (float*)d_out;  // state buffer lives in d_out (out == msg each step)

  hipMemsetAsync(cnt, 0, (size_t)NN * sizeof(float), stream);
  k_prep <<<132, 256, 0, stream>>>(We2, be2, root, W_ih, W_hh, BF, rootF, WihF, WhhF);
  k_count<<<(EE + 255) / 256, 256, 0, stream>>>(ei, cnt);
  k_proj <<<NN / 8, 256, 0, stream>>>(x, Wp, bp, h);

  for (int s = 0; s < NSTEPS; ++s) {
    hipMemsetAsync(agg, 0, (size_t)NN * DD * sizeof(float), stream);
    k_gemm<<<(NTILES * 32 + 255) / 256, 256, 0, stream>>>(h, BF, P);
    k_edge<<<EE / 8, 256, 0, stream>>>(ei, eattr, We1, be1, P, agg);
    k_node<<<(NTILES * 32 + 255) / 256, 256, 0, stream>>>(h, agg, cnt, rootF, cbias,
                                                          WihF, WhhF, b_ih, b_hh);
  }
}
